// CantorModalityFusion_46059229283214
// MI455X (gfx1250) — compile-verified, matrix operand order checked by reading, not run on hardware
//
#include <hip/hip_runtime.h>
#include <hip/hip_bf16.h>
#include <math.h>
#include <stdint.h>

typedef __bf16 bf16_t;
typedef __bf16 v16bf __attribute__((ext_vector_type(16)));
typedef float  v8f   __attribute__((ext_vector_type(8)));
typedef unsigned int u32x4 __attribute__((ext_vector_type(4)));
typedef int v4i __attribute__((ext_vector_type(4)));

union AFrag { v16bf v; u32x4 u[2]; };
union H8   { u32x4 u; bf16_t h[8]; };

static constexpr int Bc = 4, Sc = 2048, Dc = 1024, Hc = 16, HDc = 64, Mc = 4;

// ---------- CDNA5 async global->LDS staging (fallback: plain ds store) ----------
#if defined(__HIP_DEVICE_COMPILE__) && __has_builtin(__builtin_amdgcn_global_load_async_to_lds_b128)
#define USE_ASYNC_LDS 1
#else
#define USE_ASYNC_LDS 0
#endif

#if USE_ASYNC_LDS
typedef __attribute__((address_space(1))) v4i* gv4p;
typedef __attribute__((address_space(3))) v4i* lv4p;
#endif

__device__ __forceinline__ void stage16(const bf16_t* g, bf16_t* l) {
#if USE_ASYNC_LDS
  __builtin_amdgcn_global_load_async_to_lds_b128((gv4p)(uintptr_t)g, (lv4p)l, 0, 0);
#else
  *(u32x4*)l = *(const u32x4*)g;
#endif
}

__device__ __forceinline__ void wait_async0() {
#if USE_ASYNC_LDS
#if __has_builtin(__builtin_amdgcn_s_wait_asynccnt)
  __builtin_amdgcn_s_wait_asynccnt(0);
#else
  asm volatile("s_wait_asynccnt 0x0" ::: "memory");
#endif
#endif
}

// ---------------- f32 -> bf16 conversion, 8 elems/thread ----------------
__global__ void k_cvt_bf16x8(const float* __restrict__ in, bf16_t* __restrict__ out, int n8) {
  int i = blockIdx.x * blockDim.x + threadIdx.x;
  if (i >= n8) return;
  const float4* p = (const float4*)in + (size_t)i * 2;
  float4 f0 = p[0], f1 = p[1];
  H8 o;
  o.h[0] = (bf16_t)f0.x; o.h[1] = (bf16_t)f0.y; o.h[2] = (bf16_t)f0.z; o.h[3] = (bf16_t)f0.w;
  o.h[4] = (bf16_t)f1.x; o.h[5] = (bf16_t)f1.y; o.h[6] = (bf16_t)f1.z; o.h[7] = (bf16_t)f1.w;
  *((u32x4*)out + i) = o.u;
}

__device__ __forceinline__ v8f wmma_bf16(v16bf a, v16bf b, v8f c) {
  return __builtin_amdgcn_wmma_f32_16x16x32_bf16(false, a, false, b, (short)0, c, false, false);
}

// ---------------- WMMA GEMM:  C = A(MxK) * W(NxK)^T + bias + extra ----------------
// Block = 256 threads = 8 waves; block tile 256x64. Weight strip (64 x K) is shared by
// all 8 waves -> staged through LDS (double-buffered 64x32 bf16 tiles, 80B row stride,
// bank-conflict-free ds_load_b128), async-copied via GLOBAL_LOAD_ASYNC_TO_LDS_B128.
template <typename OUT_T, bool REMAP>
__global__ void k_wmma_gemm(const bf16_t* __restrict__ A,
                            const bf16_t* __restrict__ W,
                            const float* __restrict__ bias,   // len N or null
                            const float* __restrict__ extra,  // len N or null
                            OUT_T* __restrict__ C,
                            int validRows, int N, int K,
                            int m_idx, int sl) {
  constexpr int LS = 40;                 // LDS row stride in halves (32 + 8 pad)
  __shared__ bf16_t sB[2][64 * LS];

  const int tid     = threadIdx.x;
  const int lane    = tid & 31;
  const int wave    = tid >> 5;
  const int rowBase = blockIdx.y * 256 + wave * 32;
  const int colBase = blockIdx.x * 64;
  const int lrow    = lane & 15;
  const int khalf   = (lane >> 4) << 3;  // 0 | 8

  // cooperative staging assignment: thread -> (weight row, 8-half chunk)
  const int srow   = tid >> 2;           // 0..63
  const int schunk = (tid & 3) << 3;     // 0,8,16,24
  const bf16_t* wsrc = W + (size_t)(colBase + srow) * K + schunk;
  const int sdoff = srow * LS + schunk;

  v8f acc[2][4];
#pragma unroll
  for (int mt = 0; mt < 2; ++mt)
#pragma unroll
    for (int nt = 0; nt < 4; ++nt) acc[mt][nt] = (v8f)0.0f;

  int ar0 = rowBase + lrow;       if (ar0 > validRows - 1) ar0 = validRows - 1;
  int ar1 = rowBase + 16 + lrow;  if (ar1 > validRows - 1) ar1 = validRows - 1;
  const bf16_t* a0 = A + (size_t)ar0 * K;
  const bf16_t* a1 = A + (size_t)ar1 * K;

  // prime stage 0
  stage16(wsrc, &sB[0][sdoff]);

  int buf = 0;
#pragma unroll 1
  for (int k = 0; k < K; k += 32) {
    wait_async0();
    __syncthreads();
    if (k + 32 < K) stage16(wsrc + k + 32, &sB[buf ^ 1][sdoff]);

    AFrag fa0, fa1;
    fa0.u[0] = *(const u32x4*)(a0 + k + khalf);
    fa0.u[1] = *(const u32x4*)(a0 + k + 16 + khalf);
    fa1.u[0] = *(const u32x4*)(a1 + k + khalf);
    fa1.u[1] = *(const u32x4*)(a1 + k + 16 + khalf);

#pragma unroll
    for (int nt = 0; nt < 4; ++nt) {
      const bf16_t* bp = &sB[buf][(nt * 16 + lrow) * LS];
      AFrag fb;
      fb.u[0] = *(const u32x4*)(bp + khalf);
      fb.u[1] = *(const u32x4*)(bp + 16 + khalf);
      acc[0][nt] = wmma_bf16(fa0.v, fb.v, acc[0][nt]);
      acc[1][nt] = wmma_bf16(fa1.v, fb.v, acc[1][nt]);
    }
    buf ^= 1;
  }

  float badd[4];
#pragma unroll
  for (int nt = 0; nt < 4; ++nt) {
    int col = colBase + nt * 16 + lrow;
    float v = bias ? bias[col] : 0.0f;
    if (extra) v += extra[col];
    badd[nt] = v;
  }

  const int subM = (lane >> 4) << 3;  // C layout: lanes 16-31 hold M = r + 8
#pragma unroll
  for (int mt = 0; mt < 2; ++mt) {
#pragma unroll
    for (int r = 0; r < 8; ++r) {
      int row = rowBase + mt * 16 + subM + r;
      if (row < validRows) {
        size_t orow;
        if (REMAP) {
          int b = row / sl;
          int s = row - b * sl;
          orow = (size_t)(b * Mc + m_idx) * Sc + s;
        } else {
          orow = (size_t)row;
        }
#pragma unroll
        for (int nt = 0; nt < 4; ++nt) {
          int col = colBase + nt * 16 + lrow;
          C[orow * (size_t)N + col] = (OUT_T)(acc[mt][nt][r] + badd[nt]);
        }
      }
    }
  }
}

// ---------------- routed positional attention + mean over modalities ----------------
// One thread per (b, s, h); b128-vectorized bf16 reads.
__global__ void k_attn_fuse(const bf16_t* __restrict__ Q,
                            const bf16_t* __restrict__ Kb,
                            const bf16_t* __restrict__ Vb,
                            const float* __restrict__ temp,
                            bf16_t* __restrict__ fused,
                            int r0, int r1, int r2, int r3) {
  int t = blockIdx.x * blockDim.x + threadIdx.x;
  if (t >= Bc * Sc * Hc) return;
  int h  = t & (Hc - 1);
  int bs = t >> 4;
  int s  = bs & (Sc - 1);
  int b  = bs >> 11;

  const float scale = 1.0f / (8.0f * fabsf(temp[0]));  // sqrt(HD)=8
  int routes[4] = {r0, r1, r2, r3};

  float acc[HDc];
#pragma unroll
  for (int j = 0; j < HDc; ++j) acc[j] = 0.0f;

  for (int m = 0; m < Mc; ++m) {
    const bf16_t* q = Q + (((size_t)(b * Mc + m) * Sc + s) * Dc) + h * HDc;
    H8 qv[8];
#pragma unroll
    for (int c = 0; c < 8; ++c) qv[c].u = *((const u32x4*)q + c);

    float sc[3];
#pragma unroll
    for (int w = 0; w < 3; ++w) {
      int r = (routes[m] >> (8 * w)) & 0xF;
      const bf16_t* kk = Kb + (((size_t)(b * Mc + r) * Sc + s) * Dc) + h * HDc;
      float d = 0.0f;
#pragma unroll
      for (int c = 0; c < 8; ++c) {
        H8 ka; ka.u = *((const u32x4*)kk + c);
#pragma unroll
        for (int j = 0; j < 8; ++j) d += (float)qv[c].h[j] * (float)ka.h[j];
      }
      sc[w] = d * scale;
    }
    float mx = fmaxf(sc[0], fmaxf(sc[1], sc[2]));
    float e0 = __expf(sc[0] - mx), e1 = __expf(sc[1] - mx), e2 = __expf(sc[2] - mx);
    float inv = 1.0f / (e0 + e1 + e2);
    float pw[3] = {e0 * inv, e1 * inv, e2 * inv};
#pragma unroll
    for (int w = 0; w < 3; ++w) {
      int r = (routes[m] >> (8 * w)) & 0xF;
      const bf16_t* vv = Vb + (((size_t)(b * Mc + r) * Sc + s) * Dc) + h * HDc;
      float p = pw[w];
#pragma unroll
      for (int c = 0; c < 8; ++c) {
        H8 va; va.u = *((const u32x4*)vv + c);
#pragma unroll
        for (int j = 0; j < 8; ++j) acc[c * 8 + j] += p * (float)va.h[j];
      }
    }
  }

  bf16_t* o = fused + (size_t)bs * Dc + h * HDc;
#pragma unroll
  for (int c = 0; c < 8; ++c) {
    H8 ov;
#pragma unroll
    for (int j = 0; j < 8; ++j) ov.h[j] = (bf16_t)(acc[c * 8 + j] * 0.25f);
    *((u32x4*)o + c) = ov.u;
  }
}

// ---------------- host-side Cantor routes (matches reference) ----------------
static void cantor_routes(int routes[4][3]) {
  double c[4];
  for (int i = 0; i < 4; ++i) {
    double x = (double)i / 3.0;
    if (x < 1e-6) x = 1e-6;
    if (x > 1.0 - 1e-6) x = 1.0 - 1e-6;
    double val = 0.0, f = 0.5;
    for (int d = 0; d < 8; ++d) {
      x *= 3.0;
      int dd = (int)x;
      x -= dd;
      if (dd == 2) val += f;
      f *= 0.5;
    }
    c[i] = val;
  }
  for (int i = 0; i < 4; ++i) {
    int idx[4] = {0, 1, 2, 3};
    double dist[4];
    for (int j = 0; j < 4; ++j) dist[j] = fabs(c[i] - c[j]);
    for (int a = 1; a < 4; ++a) {
      int id = idx[a]; double dv = dist[id];
      int p = a - 1;
      while (p >= 0 && dist[idx[p]] > dv) { idx[p + 1] = idx[p]; --p; }
      idx[p + 1] = id;
    }
    for (int w = 0; w < 3; ++w) routes[i][w] = idx[w];
  }
}

extern "C" void kernel_launch(void* const* d_in, const int* in_sizes, int n_in,
                              void* d_out, int out_size, void* d_ws, size_t ws_size,
                              hipStream_t stream) {
  const float* x_mod[4]  = {(const float*)d_in[0], (const float*)d_in[3],
                            (const float*)d_in[6], (const float*)d_in[9]};
  const float* W_mod[4]  = {(const float*)d_in[1], (const float*)d_in[4],
                            (const float*)d_in[7], (const float*)d_in[10]};
  const float* b_mod[4]  = {(const float*)d_in[2], (const float*)d_in[5],
                            (const float*)d_in[8], (const float*)d_in[11]};
  const float* mod_emb   = (const float*)d_in[12];
  const float* Wq = (const float*)d_in[13]; const float* bq = (const float*)d_in[14];
  const float* Wk = (const float*)d_in[15]; const float* bk = (const float*)d_in[16];
  const float* Wv = (const float*)d_in[17]; const float* bv = (const float*)d_in[18];
  const float* Wo = (const float*)d_in[19]; const float* bo = (const float*)d_in[20];
  const float* temp = (const float*)d_in[21];

  const int dims[4] = {768, 1024, 512, 2048};
  const int sls[4]  = {2048, 1024, 1500, 512};

  size_t off = 0;
  auto alloc = [&](size_t bytes) -> size_t {
    size_t o = off;
    off = (off + bytes + 255) & ~(size_t)255;
    return o;
  };
  char* ws = (char*)d_ws;
  const size_t tokRows = (size_t)Bc * Mc * Sc;  // 32768
  size_t off_stacked = alloc(tokRows * Dc * 2);
  size_t off_Q       = alloc(tokRows * Dc * 2);
  size_t off_K       = alloc(tokRows * Dc * 2);
  size_t off_V       = alloc(tokRows * Dc * 2);
  size_t off_fused   = alloc((size_t)Bc * Sc * Dc * 2);
  size_t off_wq = alloc((size_t)Dc * Dc * 2);
  size_t off_wk = alloc((size_t)Dc * Dc * 2);
  size_t off_wv = alloc((size_t)Dc * Dc * 2);
  size_t off_wo = alloc((size_t)Dc * Dc * 2);
  size_t off_wm[4], off_xa[4];
  for (int m = 0; m < 4; ++m) off_wm[m] = alloc((size_t)Dc * dims[m] * 2);
  for (int m = 0; m < 4; ++m) off_xa[m] = alloc((size_t)Bc * sls[m] * dims[m] * 2);
  (void)ws_size;

  bf16_t* stacked = (bf16_t*)(ws + off_stacked);
  bf16_t* Qb = (bf16_t*)(ws + off_Q);
  bf16_t* Kb = (bf16_t*)(ws + off_K);
  bf16_t* Vb = (bf16_t*)(ws + off_V);
  bf16_t* fused = (bf16_t*)(ws + off_fused);
  bf16_t* wq = (bf16_t*)(ws + off_wq);
  bf16_t* wk = (bf16_t*)(ws + off_wk);
  bf16_t* wv = (bf16_t*)(ws + off_wv);
  bf16_t* wo = (bf16_t*)(ws + off_wo);

  auto cvt = [&](const float* src, bf16_t* dst, size_t n) {
    int n8 = (int)(n / 8);  // all sizes here are multiples of 8
    k_cvt_bf16x8<<<dim3((n8 + 255) / 256), dim3(256), 0, stream>>>(src, dst, n8);
  };

  // ---- stage 0: bf16 conversions ----
  for (int m = 0; m < 4; ++m) {
    cvt(x_mod[m], (bf16_t*)(ws + off_xa[m]), (size_t)Bc * sls[m] * dims[m]);
    cvt(W_mod[m], (bf16_t*)(ws + off_wm[m]), (size_t)Dc * dims[m]);
  }
  cvt(Wq, wq, (size_t)Dc * Dc);
  cvt(Wk, wk, (size_t)Dc * Dc);
  cvt(Wv, wv, (size_t)Dc * Dc);
  cvt(Wo, wo, (size_t)Dc * Dc);

  // ---- stage 1: input projections (scatter into zero-padded stacked buffer) ----
  (void)hipMemsetAsync(stacked, 0, tokRows * Dc * 2, stream);
  for (int m = 0; m < 4; ++m) {
    int validRows = Bc * sls[m];
    dim3 grid(Dc / 64, (unsigned)((validRows + 255) / 256));
    k_wmma_gemm<bf16_t, true><<<grid, dim3(256), 0, stream>>>(
        (const bf16_t*)(ws + off_xa[m]), (const bf16_t*)(ws + off_wm[m]),
        b_mod[m], mod_emb + (size_t)m * Dc, stacked,
        validRows, Dc, dims[m], m, sls[m]);
  }

  // ---- stage 2: Q/K/V projections over all 32768 tokens ----
  {
    dim3 grid(Dc / 64, (unsigned)(tokRows / 256));
    k_wmma_gemm<bf16_t, false><<<grid, dim3(256), 0, stream>>>(
        stacked, wq, bq, nullptr, Qb, (int)tokRows, Dc, Dc, 0, 0);
    k_wmma_gemm<bf16_t, false><<<grid, dim3(256), 0, stream>>>(
        stacked, wk, bk, nullptr, Kb, (int)tokRows, Dc, Dc, 0, 0);
    k_wmma_gemm<bf16_t, false><<<grid, dim3(256), 0, stream>>>(
        stacked, wv, bv, nullptr, Vb, (int)tokRows, Dc, Dc, 0, 0);
  }

  // ---- stage 3: routed positional attention + mean over modalities ----
  int routes[4][3];
  cantor_routes(routes);
  int rp[4];
  for (int m = 0; m < 4; ++m)
    rp[m] = routes[m][0] | (routes[m][1] << 8) | (routes[m][2] << 16);
  {
    int total = Bc * Sc * Hc;
    k_attn_fuse<<<dim3((total + 255) / 256), dim3(256), 0, stream>>>(
        Qb, Kb, Vb, temp, fused, rp[0], rp[1], rp[2], rp[3]);
  }

  // ---- stage 4: output projection -> f32 d_out ----
  {
    int rows = Bc * Sc;  // 8192
    dim3 grid(Dc / 64, (unsigned)(rows / 256));
    k_wmma_gemm<float, false><<<grid, dim3(256), 0, stream>>>(
        fused, wo, bo, nullptr, (float*)d_out, rows, Dc, Dc, 0, 0);
  }
  (void)in_sizes; (void)n_in; (void)out_size;
}
